// nHFC_53257594470997
// MI455X (gfx1250) — matrix operand order repaired and checked by reading, not compiled, not run on hardware
//
#include <hip/hip_runtime.h>
#include <hip/hip_bf16.h>
#include <math.h>

// ---------------------------------------------------------------------------
// CDNA5 (gfx1250, wave32) SchNet stack.
// All GEMMs on v_wmma_f32_16x16x32_bf16 (bf16 A/B, f32 accumulate).
// B matrices pre-packed into WMMA fragment order (32B contiguous per lane);
// A tiles staged in LDS in fragment order -> b128 loads, no per-elem guards.
// ---------------------------------------------------------------------------

typedef __attribute__((ext_vector_type(16))) __bf16 v16bf;
typedef __attribute__((ext_vector_type(8)))  float  v8f;

#define NNODES 20000
#define NEDGES 640000
#define GAUSS  10
#define ETILE  32                  // edges per workgroup in edge kernel
#define LOG2F_ 0.6931471805599453f
#define DELTA_ (10.0f / 9.0f)      // CUTOFF/(G-1)
#define COEFF_ (-0.405f)           // -0.5/DELTA_^2

__device__ __forceinline__ float sspf(float x) {
    float sp = (x > 20.0f) ? x : __logf(1.0f + __expf(x));
    return sp - LOG2F_;
}

__device__ __forceinline__ v8f wmma_bf16(v16bf a, v16bf b, v8f c) {
    return __builtin_amdgcn_wmma_f32_16x16x32_bf16(
        false, a, false, b, (short)0, c, false, false);
}

// ---------------------------------------------------------------------------
// Pack f32 weight W[K x Nc] into bf16 WMMA-B fragment order.
// Tile (kb, nb) is a 32x16 K*N block stored as 512 bf16 = 1 KB:
//   tile index = nb*nkb + kb (K-contiguous streaming in the GEMM inner loop)
//   within tile: lane = {n & 15, khalf}, 16 elems/lane contiguous (32 B)
//   elem e of lane -> k = kb*32 + (e & 7) + (lane>=16 ? 8 : 0) + (e>=8 ? 16 : 0), zero-pad k >= K
// ---------------------------------------------------------------------------
__global__ void pack_b_kernel(const float* __restrict__ W, __bf16* __restrict__ P,
                              int K, int Nc, int nkb) {
    int p = blockIdx.x * blockDim.x + threadIdx.x;
    if (p >= (nkb << 5) * Nc) return;
    int tile = p >> 9, within = p & 511;
    int nb = tile / nkb, kb = tile - nb * nkb;
    int lane = within >> 4, e = within & 15;
    int n = (nb << 4) + (lane & 15);
    int k = (kb << 5) + (e & 7) + ((lane >> 4) << 3) + ((e >> 3) << 4);
    float v = (k < K) ? W[(size_t)k * Nc + n] : 0.0f;
    P[p] = (__bf16)v;
}

// o[m, 0:C] = a[m, oa:oa+C] * b[m, ob:ob+C]
__global__ void ewmul_kernel(const float* __restrict__ a, int sa, int oa,
                             const float* __restrict__ b, int sb, int ob,
                             float* __restrict__ o, int M, int C) {
    int i = blockIdx.x * blockDim.x + threadIdx.x;
    if (i >= M * C) return;
    int m = i / C, c = i - m * C;
    o[(size_t)m * C + c] = a[(size_t)m * sa + oa + c] * b[(size_t)m * sb + ob + c];
}

// ---------------------------------------------------------------------------
// Generic GEMM: C[M,Nc] = act(A_f32[:, aoff..aoff+K) @ Bpacked + bias)
// 128 threads = 4 waves; each wave: one 16-row M block x two 16-col N blocks.
// A tile staged f32->bf16 into LDS in fragment order.
// M and all dims are multiples of 16 -> no fine-grained guards.
// ---------------------------------------------------------------------------
template <bool ACT>
__global__ void gemm_kernel(const float* __restrict__ A, int lda, int aoff, int K,
                            const __bf16* __restrict__ Bp, int nkb,
                            const float* __restrict__ bias,
                            float* __restrict__ C, int Nc) {
    __shared__ __align__(32) __bf16 sAf[512];
    const int tid  = threadIdx.x;
    const int wave = tid >> 5;
    const int lane = tid & 31;
    const int hi   = lane >> 4;
    const int lx   = lane & 15;
    const int m0   = blockIdx.x * 16;
    const int nbtot = Nc >> 4;
    const int nb0   = (blockIdx.y * 4 + wave) * 2;

    // Staging map (m, kin) -> fragment slot; each thread owns 4 consecutive k.
    const int sm   = tid >> 3;
    const int skin = (tid & 7) << 2;
    const int sidx = ((sm + (((skin >> 3) & 1) << 4)) << 4) +
                     (skin & 7) + ((skin >> 4) << 3);
    const float* Arow = A + (size_t)(m0 + sm) * lda + aoff + skin;
    const v16bf* afp  = (const v16bf*)sAf;

    v8f acc0 = {}, acc1 = {};
    for (int kb = 0; kb < nkb; ++kb) {
        const int k0 = kb << 5;
        __syncthreads();
        float4 v4 = make_float4(0.f, 0.f, 0.f, 0.f);
        if (k0 + skin < K) v4 = *(const float4*)(Arow + k0);  // coalesced b128
        sAf[sidx + 0] = (__bf16)v4.x;
        sAf[sidx + 1] = (__bf16)v4.y;
        sAf[sidx + 2] = (__bf16)v4.z;
        sAf[sidx + 3] = (__bf16)v4.w;
        __syncthreads();
        v16bf af = afp[lane];                                  // 2x ds_load_b128
        if (nb0 < nbtot) {
            const v16bf* bp = (const v16bf*)(Bp + (((size_t)nb0 * nkb + kb) << 9));
            acc0 = wmma_bf16(af, bp[lane], acc0);              // 2x global b128
        }
        if (nb0 + 1 < nbtot) {
            const v16bf* bp = (const v16bf*)(Bp + (((size_t)(nb0 + 1) * nkb + kb) << 9));
            acc1 = wmma_bf16(af, bp[lane], acc1);
        }
    }
#pragma unroll
    for (int j = 0; j < 2; ++j) {
        int nb = nb0 + j;
        if (nb >= nbtot) break;
        v8f acc = j ? acc1 : acc0;
        int n = (nb << 4) + lx;
        float bv = bias ? bias[n] : 0.0f;
#pragma unroll
        for (int r = 0; r < 8; ++r) {
            int m = m0 + r + hi * 8;
            float v = acc[r] + bv;
            if (ACT) v = sspf(v);
            C[(size_t)m * Nc + n] = v;
        }
    }
}

// ---------------------------------------------------------------------------
// Fused edge pipeline: rbf -> filt1 (K=10, VALU) -> WMMA vs packed mw2
//   -> msg = xw[src]*filt -> atomicAdd agg[dst].
// 32 edges / WG, 8 waves = 2 edge-row-blocks x 4 column stripes.
// filt1 kept in LDS in fragment order; the E x D filt tensor never hits HBM.
// ---------------------------------------------------------------------------
__global__ void edge_kernel(const float* __restrict__ pos,
                            const int* __restrict__ srcI,
                            const int* __restrict__ dstI,
                            const float* __restrict__ mw1,
                            const float* __restrict__ mb1,
                            const __bf16* __restrict__ mw2p,
                            const float* __restrict__ mb2,
                            const float* __restrict__ xw,
                            float* __restrict__ agg,
                            int D, int Dpad, int dsh, int nkb) {
    __shared__ float rbf[ETILE * GAUSS];
    __shared__ __align__(32) __bf16 f1[ETILE * 512];   // fragment order
    const int tid = threadIdx.x;                        // 0..255
    const int e0  = blockIdx.x * ETILE;

    // Stage 1: per-edge distance + Gaussian RBF (E % ETILE == 0, no guards)
    if (tid < ETILE) {
        int e = e0 + tid;
        int s = srcI[e], dd = dstI[e];
        float dx = pos[dd * 3 + 0] - pos[s * 3 + 0];
        float dy = pos[dd * 3 + 1] - pos[s * 3 + 1];
        float dz = pos[dd * 3 + 2] - pos[s * 3 + 2];
        float dist = sqrtf(dx * dx + dy * dy + dz * dz);
#pragma unroll
        for (int g = 0; g < GAUSS; ++g) {
            float t = dist - (float)g * DELTA_;
            rbf[tid * GAUSS + g] = __expf(COEFF_ * t * t);
        }
    }
    __syncthreads();

    // Stage 2: filt1 = ssp(rbf @ mw1 + mb1) -> bf16 LDS (fragment order),
    // zero-padded to Dpad so the WMMA K-loop needs no guards.
    for (int base = tid << 2; base < ETILE * Dpad; base += 1024) {
        int m = base >> dsh;
        int k = base & (Dpad - 1);      // multiple of 4
        float4 a4 = make_float4(0.f, 0.f, 0.f, 0.f);
        if (k < D) {
            a4 = *(const float4*)(mb1 + k);
#pragma unroll
            for (int g = 0; g < GAUSS; ++g) {
                float  r  = rbf[m * GAUSS + g];
                float4 wv = *(const float4*)(mw1 + g * D + k);
                a4.x += r * wv.x; a4.y += r * wv.y;
                a4.z += r * wv.z; a4.w += r * wv.w;
            }
            a4.x = sspf(a4.x); a4.y = sspf(a4.y);
            a4.z = sspf(a4.z); a4.w = sspf(a4.w);
        }
        int kin = k & 31, kb = k >> 5, rb = m >> 4;
        int lane = (m & 15) + (((kin >> 3) & 1) << 4);
        int ee   = (kin & 7) + ((kin >> 4) << 3);       // 4 consecutive slots
        __bf16* dst = &f1[((rb * nkb + kb) << 9) + (lane << 4) + ee];
        dst[0] = (__bf16)a4.x; dst[1] = (__bf16)a4.y;
        dst[2] = (__bf16)a4.z; dst[3] = (__bf16)a4.w;
    }
    __syncthreads();

    // Stage 3: filt = filt1 @ mw2 + mb2 via WMMA, fused gather/scale/scatter.
    const int wave = tid >> 5, lane = tid & 31, hi = lane >> 4, lx = lane & 15;
    const int rb = wave & 1;            // edge-row block (16 edges)
    const int nbtot = D >> 4;
    for (int nb = wave >> 1; nb < nbtot; nb += 4) {
        v8f acc = {};
        for (int kb = 0; kb < nkb; ++kb) {
            v16bf af = *(const v16bf*)&f1[((rb * nkb + kb) << 9) + (lane << 4)];
            const v16bf* bp = (const v16bf*)(mw2p + (((size_t)nb * nkb + kb) << 9));
            acc = wmma_bf16(af, bp[lane], acc);
        }
        int n = (nb << 4) + lx;
        float bv = mb2[n];
#pragma unroll
        for (int r = 0; r < 8; ++r) {
            int e = e0 + rb * 16 + r + hi * 8;
            int s = srcI[e], dd = dstI[e];
            float f = acc[r] + bv;
            atomicAdd(&agg[(size_t)dd * D + n], xw[(size_t)s * D + n] * f);
        }
    }
}

// ---------------------------------------------------------------------------
// Host-side orchestration
// ---------------------------------------------------------------------------
struct LayerP {
    const float *b2, *b3, *mb1, *mb2, *mw1;
    const float *mw2f, *w1f, *w2f, *w3f;
    __bf16 *mw2, *w1, *w2, *w3;
    int din, dout;
};

extern "C" void kernel_launch(void* const* d_in, const int* in_sizes, int n_in,
                              void* d_out, int out_size, void* d_ws, size_t ws_size,
                              hipStream_t stream) {
    (void)in_sizes; (void)n_in; (void)out_size; (void)ws_size;
    const float* x3d = (const float*)d_in[1];
    const float* pos = (const float*)d_in[2];
    const int*   eix = (const int*)d_in[4];       // [2, E] int32
    const int* srcI = eix;
    const int* dstI = eix + NEDGES;

    // Sorted pytree order: dwconv(5), proj_in(14), proj_out(23), pws(32,41,50,59)
    // per layer: b2,b3,mb1,mb2,mw1,mw2,w1,w2,w3
    auto getL = [&](int b, int din, int dout) {
        LayerP L;
        L.b2  = (const float*)d_in[b + 0]; L.b3  = (const float*)d_in[b + 1];
        L.mb1 = (const float*)d_in[b + 2]; L.mb2 = (const float*)d_in[b + 3];
        L.mw1 = (const float*)d_in[b + 4]; L.mw2f = (const float*)d_in[b + 5];
        L.w1f = (const float*)d_in[b + 6]; L.w2f  = (const float*)d_in[b + 7];
        L.w3f = (const float*)d_in[b + 8];
        L.mw2 = L.w1 = L.w2 = L.w3 = nullptr;
        L.din = din; L.dout = dout;
        return L;
    };
    LayerP Ldw  = getL(5, 496, 496);
    LayerP Lin  = getL(14, 256, 512);
    LayerP Lout = getL(23, 256, 256);
    LayerP Lpw[4] = { getL(32, 16, 32), getL(41, 32, 64),
                      getL(50, 64, 128), getL(59, 128, 256) };

    char* ws = (char*)d_ws;
    auto carve = [&](size_t bytes) {
        char* p = ws; ws += (bytes + 255) & ~(size_t)255; return p;
    };
    float* fused = (float*)carve((size_t)NNODES * 512 * 4);
    float* dw    = (float*)carve((size_t)NNODES * 496 * 4);
    float* xw    = (float*)carve((size_t)NNODES * 512 * 4);
    float* agg   = (float*)carve((size_t)NNODES * 512 * 4);
    float* mid   = (float*)carve((size_t)NNODES * 512 * 4);
    float* hA    = (float*)carve((size_t)NNODES * 256 * 4);
    float* hB    = (float*)carve((size_t)NNODES * 256 * 4);

    auto pack = [&](const float* W, int K, int Nc) -> __bf16* {
        int nkb = (K + 31) >> 5;
        size_t n = (size_t)(nkb << 5) * Nc;
        __bf16* p = (__bf16*)carve(n * 2);
        pack_b_kernel<<<dim3((unsigned)((n + 255) / 256)), 256, 0, stream>>>(
            W, p, K, Nc, nkb);
        return p;
    };
    auto prep = [&](LayerP& L) {
        L.mw2 = pack(L.mw2f, L.dout, L.dout);
        L.w1  = pack(L.w1f,  L.din,  L.dout);
        L.w2  = pack(L.w2f,  L.dout, L.dout);
        L.w3  = pack(L.w3f,  L.dout, L.dout);
    };
    prep(Ldw); prep(Lin); prep(Lout);
    for (int i = 0; i < 4; ++i) prep(Lpw[i]);

    auto run_layer = [&](LayerP& L, const float* Ain, int lda, int aoff, float* out) {
        const int K = L.din, D = L.dout;
        const int nkbK = (K + 31) >> 5, nkbD = (D + 31) >> 5;
        const int Dpad = nkbD << 5;
        const int dsh  = __builtin_ctz(Dpad);
        dim3 g((NNODES + 15) / 16, ((D >> 4) + 7) / 8);
        // 1) xw = Ain @ w1
        gemm_kernel<false><<<g, 128, 0, stream>>>(Ain, lda, aoff, K, L.w1, nkbK,
                                                  nullptr, xw, D);
        // 2) agg = 0; fused edge message + scatter
        hipMemsetAsync(agg, 0, (size_t)NNODES * D * 4, stream);
        edge_kernel<<<dim3(NEDGES / ETILE), 256, 0, stream>>>(
            pos, srcI, dstI, L.mw1, L.mb1, L.mw2, L.mb2, xw, agg, D, Dpad, dsh, nkbD);
        // 3) mid = ssp(agg @ w2 + b2)
        gemm_kernel<true><<<g, 128, 0, stream>>>(agg, D, 0, D, L.w2, nkbD,
                                                 L.b2, mid, D);
        // 4) out = mid @ w3 + b3
        gemm_kernel<false><<<g, 128, 0, stream>>>(mid, D, 0, D, L.w3, nkbD,
                                                  L.b3, out, D);
    };

    run_layer(Lin, x3d, 256, 0, fused);                 // fused = schnet(proj_in, x_3d)
    run_layer(Ldw, fused, 512, 16, dw);                 // dw = schnet(dwconv, fused[:,16:])
    ewmul_kernel<<<(NNODES * 16 + 255) / 256, 256, 0, stream>>>(
        fused, 512, 0, dw, 496, 0, hA, NNODES, 16);     // h = pwa * dw[:, :16]

    const int dims[5] = {16, 32, 64, 128, 256};
    const int offs[5] = {0, 16, 48, 112, 240};
    float* cur = hA;
    for (int i = 0; i < 4; ++i) {
        run_layer(Lpw[i], cur, dims[i], 0, hB);
        int C = dims[i + 1];
        ewmul_kernel<<<(NNODES * C + 255) / 256, 256, 0, stream>>>(
            hB, C, 0, dw, 496, offs[i + 1], hA, NNODES, C);
        cur = hA;
    }
    run_layer(Lout, cur, 256, 0, (float*)d_out);        // out = schnet(proj_out, h)
}
